// DifferentialAttentionLayer_54932631715887
// MI455X (gfx1250) — matrix operand order repaired and checked by reading, not compiled, not run on hardware
//
#include <hip/hip_runtime.h>
#include <hip/hip_bf16.h>

// ---------------------------------------------------------------------------
// Differential attention (B=2, L=1024, D=1024, H=16, HD=32) for gfx1250.
// GEMMs via v_wmma_f32_16x16x32_bf16 (fp32 accumulate), softmax/prior in fp32,
// TDM (tensor_load_to_lds) staging of V tiles with LDS row padding.
// ---------------------------------------------------------------------------

typedef __attribute__((ext_vector_type(16))) __bf16 v16bf;
typedef __attribute__((ext_vector_type(8)))  float  v8f;
typedef __attribute__((ext_vector_type(4)))  unsigned v4u;
typedef __attribute__((ext_vector_type(8)))  unsigned v8u;

#define B_  2
#define L_  1024
#define D_  1024
#define H_  16
#define HD_ 32

__device__ __constant__ float kLambdaInit  = 0.35550906759096926f;   // 0.8-0.6*exp(-0.3)
__device__ __constant__ float kScale       = 0.17677669529663687f;   // 32^-0.5
__device__ __constant__ float kInvSqrt2Pi  = 0.3989422804014327f;
__device__ __constant__ float kLn3         = 1.0986122886681098f;

union BF16Frag { v16bf v; __bf16 e[16]; };

__device__ __forceinline__ void cvt4(__bf16* h, float4 a) {
  h[0] = (__bf16)a.x; h[1] = (__bf16)a.y; h[2] = (__bf16)a.z; h[3] = (__bf16)a.w;
}

// A-matrix 16x32 bf16 fragment: lane holds row (lane&15); element e is
// K = kb + e, kb = (lane>>4)*16.
// Caller passes p pointing at (row, kb); we read p[0..15].
__device__ __forceinline__ void loadA_frag(BF16Frag& f, const float* p) {
  const float4* q0 = (const float4*)p;
  cvt4(f.e + 0, q0[0]); cvt4(f.e + 4, q0[1]);
  cvt4(f.e + 8, q0[2]); cvt4(f.e + 12, q0[3]);
}

// B-matrix 32x16 bf16 fragment: lane holds col (lane&15); element e is
// K = (lane>>4)*16 + e. Caller passes p at (col, kb2); 16 contiguous floats.
__device__ __forceinline__ void loadB_frag(BF16Frag& f, const float* p) {
  const float4* q0 = (const float4*)p;
  cvt4(f.e + 0, q0[0]); cvt4(f.e + 4, q0[1]);
  cvt4(f.e + 8, q0[2]); cvt4(f.e + 12, q0[3]);
}

__device__ __forceinline__ v8f wmma_bf16(const BF16Frag& a, const BF16Frag& b, v8f c) {
  return __builtin_amdgcn_wmma_f32_16x16x32_bf16(false, a.v, false, b.v,
                                                 (short)0, c, false, false);
}

// ---------------------------------------------------------------------------
// TDM: DMA a [rows=32, cols=64] fp32 tile (row stride 64 floats in memory)
// into LDS with 1-dword padding every 64 dwords (=> 65-float LDS row pitch,
// conflict-free column reads afterwards). D# per CDNA5 ISA ch.8.
// Issued by one wave; EXEC ignored by TDM; tracked with TENSORcnt.
// ---------------------------------------------------------------------------
__device__ __forceinline__ void tdm_load_v_tile(const float* gsrc,
                                                unsigned lds_byte_off,
                                                unsigned rows_avail) {
  unsigned long long ga = (unsigned long long)(size_t)gsrc;
  v4u g0;
  g0[0] = 1u;                                   // count=1 (valid), user mode
  g0[1] = lds_byte_off;                         // lds_addr (bytes)
  g0[2] = (unsigned)ga;                         // global_addr[31:0]
  g0[3] = (unsigned)(ga >> 32) | (2u << 30);    // global_addr[56:32] | type=2
  v8u g1;
  g1[0] = (2u << 16)                            // data_size = 4 bytes
        | (1u << 20)                            // pad_enable
        | (5u << 22);                           // pad_interval: every 64 dwords
                                                // pad_amount=0 -> 1 dword pad
  g1[1] = (64u & 0xFFFFu) << 16;                // tensor_dim0 = 64 (lo16)
  g1[2] = (rows_avail & 0xFFFFu) << 16;         // dim0 hi16=0 | tensor_dim1 lo16
  g1[3] = (rows_avail >> 16) | (64u << 16);     // tensor_dim1 hi16 | tile_dim0=64
  g1[4] = 32u;                                  // tile_dim1 = 32 | tile_dim2 = 0
  g1[5] = 64u;                                  // tensor_dim0_stride = 64 (lo32)
  g1[6] = 0u;                                   // stride0 hi | stride1 lo (unused)
  g1[7] = 0u;
  asm volatile("tensor_load_to_lds %0, %1" :: "s"(g0), "s"(g1) : "memory");
}

// ---------------------------------------------------------------------------
// Generic C[M,N] = A[M,K] * W[N,K]^T (+bias), one 16x16 tile per wave.
// mode: 0 = row-major [M,N]; 1 = q layout [B,2H,L,HD];
//       2 = k layout [B,H,L,HD]; 3 = v layout [B,H/2,L,2HD]
// ---------------------------------------------------------------------------
__global__ void __launch_bounds__(128)
proj_gemm_kernel(const float* __restrict__ A, const float* __restrict__ W,
                 const float* __restrict__ bias, float* __restrict__ C,
                 int M, int N, int K, int mode) {
  const int lane = threadIdx.x & 31;
  const int wave = threadIdx.x >> 5;
  const int tm = blockIdx.x;
  const int tn = blockIdx.y * 4 + wave;
  if (tn * 16 >= N) return;                      // uniform per wave

  const int ar  = tm * 16 + (lane & 15);
  const int wr  = tn * 16 + (lane & 15);
  const int kbA = (lane >> 4) * 16;
  const int kbB = (lane >> 4) * 16;
  const float* ap = A + (size_t)ar * K + kbA;
  const float* wp = W + (size_t)wr * K + kbB;

  v8f c = {};
  for (int kt = 0; kt < K; kt += 32) {
    BF16Frag a, b;
    loadA_frag(a, ap + kt);
    loadB_frag(b, wp + kt);
    if (kt + 32 < K) {
      __builtin_prefetch(ap + kt + 32, 0, 0);    // global_prefetch_b8
      __builtin_prefetch(wp + kt + 32, 0, 0);
    }
    c = wmma_bf16(a, b, c);
  }

#pragma unroll
  for (int i = 0; i < 8; i++) {
    int m = tm * 16 + i + ((lane >> 4) << 3);
    int n = tn * 16 + (lane & 15);
    float val = c[i] + (bias ? bias[n] : 0.0f);
    size_t idx;
    if (mode == 0) {
      idx = (size_t)m * N + n;
    } else if (mode == 1) {          // q: [B, 2H, L, 32]
      int b0 = m >> 10, l = m & 1023, hh = n >> 5, d = n & 31;
      idx = (((size_t)(b0 * 32 + hh) * L_ + l) << 5) + d;
    } else if (mode == 2) {          // k: [B, H, L, 32]
      int b0 = m >> 10, l = m & 1023, hh = n >> 5, d = n & 31;
      idx = (((size_t)(b0 * 16 + hh) * L_ + l) << 5) + d;
    } else {                         // v: [B, H/2, L, 64]
      int b0 = m >> 10, l = m & 1023, hh = n >> 6, d = n & 63;
      idx = (((size_t)(b0 * 8 + hh) * L_ + l) << 6) + d;
    }
    C[idx] = val;
  }
}

// ---------------------------------------------------------------------------
__global__ void lam_kernel(const float* __restrict__ lq1, const float* __restrict__ lk1,
                           const float* __restrict__ lq2, const float* __restrict__ lk2,
                           float* __restrict__ lam_out) {
  if (threadIdx.x == 0) {
    float s1 = 0.f, s2 = 0.f;
    for (int i = 0; i < HD_; i++) { s1 += lq1[i] * lk1[i]; s2 += lq2[i] * lk2[i]; }
    lam_out[0] = __expf(s1) - __expf(s2) + kLambdaInit;
  }
}

// ---------------------------------------------------------------------------
// One block = one (b, head-pair h, 16-row query tile).
// LDS: S0[16][1024] + S1[16][1024] fp32 scores, Vt[16][64],
//      2x TDM stage buffers [32][65] (padded pitch). Total ~148 KB (<320 KB).
// ---------------------------------------------------------------------------
__global__ void __launch_bounds__(256)
diff_attn_kernel(const float* __restrict__ qws, const float* __restrict__ kws,
                 const float* __restrict__ vws, const float* __restrict__ lam_p,
                 const float* __restrict__ g, float* __restrict__ series_out,
                 float* __restrict__ vn_ws) {
  extern __shared__ float smem[];
  float* S0   = smem;                    // 16*1024
  float* S1   = smem + 16 * L_;          // 16*1024
  float* Vt   = smem + 32 * L_;          // 16*64
  float* Vst0 = Vt + 16 * 64;            // 32*65 (TDM-padded pitch)
  float* Vst1 = Vst0 + 32 * 65;          // 32*65

  const int tid = threadIdx.x, lane = tid & 31, wave = tid >> 5;
  const int qt = blockIdx.x, h = blockIdx.y, b = blockIdx.z;
  const float lam = lam_p[0];
  const float NINF = -__builtin_inff();

  // ---- phase 1: raw scores for both paired heads, causal-masked, scaled ----
  {
    const int qrow = qt * 16 + (lane & 15);
    const int kb   = (lane >> 4) * 16;
    BF16Frag a0, a1;
    loadA_frag(a0, qws + (((size_t)(b * 32 + 2 * h    ) * L_ + qrow) << 5) + kb);
    loadA_frag(a1, qws + (((size_t)(b * 32 + 2 * h + 1) * L_ + qrow) << 5) + kb);
    const int hdb = (lane >> 4) * 16;
    for (int jt = wave; jt < L_ / 16; jt += 8) {
      const int j = jt * 16 + (lane & 15);
      BF16Frag bk;
      loadB_frag(bk, kws + (((size_t)(b * 16 + h) * L_ + j) << 5) + hdb);
      v8f c0 = {}, c1 = {};
      c0 = wmma_bf16(a0, bk, c0);
      c1 = wmma_bf16(a1, bk, c1);
      const int col = jt * 16 + (lane & 15);
#pragma unroll
      for (int i = 0; i < 8; i++) {
        int r = i + ((lane >> 4) << 3);
        bool masked = col > (qt * 16 + r);
        S0[r * L_ + col] = masked ? NINF : c0[i] * kScale;
        S1[r * L_ + col] = masked ? NINF : c1[i] * kScale;
      }
    }
  }
  __syncthreads();

  // ---- phase 2: dual softmax, lambda subtraction, series softmax ----------
  {
    const int row = tid >> 4, sub = tid & 15;
    float* r0 = S0 + row * L_;
    float* r1 = S1 + row * L_;
    float mx0 = NINF, mx1 = NINF;
    for (int j = sub; j < L_; j += 16) {
      mx0 = fmaxf(mx0, r0[j]); mx1 = fmaxf(mx1, r1[j]);
    }
    for (int m = 8; m; m >>= 1) {
      mx0 = fmaxf(mx0, __shfl_xor(mx0, m));
      mx1 = fmaxf(mx1, __shfl_xor(mx1, m));
    }
    float s0 = 0.f, s1 = 0.f;
    for (int j = sub; j < L_; j += 16) {
      s0 += __expf(r0[j] - mx0); s1 += __expf(r1[j] - mx1);
    }
    for (int m = 8; m; m >>= 1) { s0 += __shfl_xor(s0, m); s1 += __shfl_xor(s1, m); }
    const float i0 = 1.f / s0, i1 = 1.f / s1;

    float awmax = NINF;
    for (int j = sub; j < L_; j += 16) {
      float aw = __expf(r0[j] - mx0) * i0 - lam * __expf(r1[j] - mx1) * i1;
      r0[j] = aw;                                   // keep aw for V aggregation
      awmax = fmaxf(awmax, aw);
    }
    for (int m = 8; m; m >>= 1) awmax = fmaxf(awmax, __shfl_xor(awmax, m));
    float ssum = 0.f;
    for (int j = sub; j < L_; j += 16) {
      float e = __expf(r0[j] - awmax);
      r1[j] = e;
      ssum += e;
    }
    for (int m = 8; m; m >>= 1) ssum += __shfl_xor(ssum, m);
    const float si = 1.f / ssum;
    const size_t sbase = ((size_t)(b * H_ + h) * L_ + qt * 16 + row) * (size_t)L_;
    for (int j = sub; j < L_; j += 16) series_out[sbase + j] = r1[j] * si;
  }
  __syncthreads();

  // ---- phase 3: V = aw @ v. TDM double-buffers 32x64 v tiles into LDS -----
  {
    const float* vbase = vws + ((size_t)(b * 8 + (h >> 1)) * L_) * 64;

    if (wave == 7) {                      // producer wave: prime buffer 0
      tdm_load_v_tile(vbase, (unsigned)(size_t)(void*)Vst0, L_);
      __builtin_amdgcn_s_wait_tensorcnt(0);
    }
    __syncthreads();

    const int nt   = wave & 3;
    const int kbA  = (lane >> 4) * 16;
    const int dcol = nt * 16 + (lane & 15);
    const int r    = lane & 15;
    const int krow = (lane >> 4) * 16;    // local K row base within stage tile
    v8f c = {};

    for (int kt = 0; kt < L_; kt += 32) {
      float* cur = ((kt >> 5) & 1) ? Vst1 : Vst0;
      float* nxt = ((kt >> 5) & 1) ? Vst0 : Vst1;
      if (wave == 7 && kt + 32 < L_) {
        tdm_load_v_tile(vbase + (size_t)(kt + 32) * 64,
                        (unsigned)(size_t)(void*)nxt, (unsigned)(L_ - kt - 32));
      }
      if (wave < 4) {
        BF16Frag a, bb;
        loadA_frag(a, S0 + r * L_ + kt + kbA);          // ds_load_b128 from LDS
#pragma unroll
        for (int e = 0; e < 16; e++)
          bb.e[e] = (__bf16)cur[(krow + e) * 65 + dcol]; // padded pitch: no bank dup
        c = wmma_bf16(a, bb, c);
      }
      if (wave == 7 && kt + 32 < L_) __builtin_amdgcn_s_wait_tensorcnt(0);
      __syncthreads();
    }
    if (wave < 4) {
#pragma unroll
      for (int i = 0; i < 8; i++) {
        int r2 = i + ((lane >> 4) << 3);
        Vt[r2 * 64 + dcol] = c[i];
      }
    }
  }
  __syncthreads();

  // ---- phase 4: RMSNorm + (1-lambda_init) scaling, write Vn ---------------
  {
    const int row = tid >> 4, sub = tid & 15;
    float vv[4];
#pragma unroll
    for (int d = 0; d < 4; d++) vv[d] = Vt[row * 64 + sub * 4 + d];
    float ssq = vv[0]*vv[0] + vv[1]*vv[1] + vv[2]*vv[2] + vv[3]*vv[3];
    for (int m = 8; m; m >>= 1) ssq += __shfl_xor(ssq, m);
    float sc = rsqrtf(ssq * (1.f / 64.f) + 1e-5f) * (1.f - kLambdaInit);
    const size_t obase = ((size_t)(b * L_ + qt * 16 + row)) * D_ + h * 64 + sub * 4;
#pragma unroll
    for (int d = 0; d < 4; d++) vn_ws[obase + d] = vv[d] * sc * g[sub * 4 + d];
  }
}

// ---------------------------------------------------------------------------
// Gaussian prior + s output (pure bandwidth + v_exp_f32).
// ---------------------------------------------------------------------------
__global__ void __launch_bounds__(256)
prior_kernel(const float* __restrict__ sigma_ws, float* __restrict__ prior_out,
             float* __restrict__ s_out) {
  const int i = blockIdx.x, h = blockIdx.y, b = blockIdx.z;
  const float sig = sigma_ws[(size_t)(b * L_ + i) * H_ + h];
  const float sv  = 1.f / (1.f + __expf(-5.f * sig)) + 1e-5f;
  const float s   = __expf(sv * kLn3) - 1.f;
  if (threadIdx.x == 0) s_out[(size_t)(b * H_ + h) * L_ + i] = s;
  const float inv2s2 = -0.5f / (s * s);
  const float coef   = kInvSqrt2Pi / s;
  const size_t base  = ((size_t)(b * H_ + h) * L_ + i) * (size_t)L_;
  const float fi = (float)i;
  for (int j = threadIdx.x; j < L_; j += 256) {
    float d = fi - (float)j;
    prior_out[base + j] = coef * __expf(d * d * inv2s2);
  }
}

// ---------------------------------------------------------------------------
extern "C" void kernel_launch(void* const* d_in, const int* in_sizes, int n_in,
                              void* d_out, int out_size, void* d_ws, size_t ws_size,
                              hipStream_t stream) {
  (void)in_sizes; (void)n_in; (void)out_size; (void)ws_size;
  const float* queries = (const float*)d_in[0];
  const float* keys    = (const float*)d_in[1];
  const float* values  = (const float*)d_in[2];
  // d_in[3] = attn_mask (bool): causal mask is known statically, unused.
  const float* Wq   = (const float*)d_in[4];
  const float* Wk   = (const float*)d_in[5];
  const float* Wv   = (const float*)d_in[6];
  const float* Wo   = (const float*)d_in[7];
  const float* Wsig = (const float*)d_in[8];
  const float* bsig = (const float*)d_in[9];
  const float* lq1  = (const float*)d_in[10];
  const float* lk1  = (const float*)d_in[11];
  const float* lq2  = (const float*)d_in[12];
  const float* lk2  = (const float*)d_in[13];
  const float* g    = (const float*)d_in[14];

  float* out = (float*)d_out;
  float* ws  = (float*)d_ws;

  // workspace layout (floats)
  float* q_ws   = ws;                 // [B,2H,L,32]  2,097,152
  float* k_ws   = ws + 2097152;       // [B,H,L,32]   1,048,576
  float* v_ws   = ws + 3145728;       // [B,H/2,L,64] 1,048,576
  float* sig_ws = ws + 4194304;       // [B,L,H]         32,768
  float* lam_ws = ws + 4227072;       // scalar (padded)
  float* vn_ws  = ws + 4227088;       // [B,L,D]      2,097,152

  // output layout (floats, concatenated in return order)
  float* o_out    = out;              // [B,L,D]      2,097,152
  float* o_series = out + 2097152;    // [B,H,L,L]   33,554,432
  float* o_prior  = out + 35651584;   // [B,H,L,L]   33,554,432
  float* o_s      = out + 69206016;   // [B,H,L]         32,768

  const int M = B_ * L_;              // 2048

  proj_gemm_kernel<<<dim3(M / 16, 16), 128, 0, stream>>>(queries, Wq,   nullptr, q_ws,   M, 1024, D_, 1);
  proj_gemm_kernel<<<dim3(M / 16, 8),  128, 0, stream>>>(keys,    Wk,   nullptr, k_ws,   M, 512,  D_, 2);
  proj_gemm_kernel<<<dim3(M / 16, 8),  128, 0, stream>>>(values,  Wv,   nullptr, v_ws,   M, 512,  D_, 3);
  proj_gemm_kernel<<<dim3(M / 16, 1),  128, 0, stream>>>(queries, Wsig, bsig,    sig_ws, M, 16,   D_, 0);
  lam_kernel<<<1, 32, 0, stream>>>(lq1, lk1, lq2, lk2, lam_ws);

  const size_t smem_bytes =
      (size_t)(32 * L_ + 16 * 64 + 2 * 32 * 65) * sizeof(float);   // 151,808 B
  hipFuncSetAttribute(reinterpret_cast<const void*>(diff_attn_kernel),
                      hipFuncAttributeMaxDynamicSharedMemorySize, (int)smem_bytes);
  diff_attn_kernel<<<dim3(L_ / 16, H_, B_), 256, smem_bytes, stream>>>(
      q_ws, k_ws, v_ws, lam_ws, g, o_series, vn_ws);

  prior_kernel<<<dim3(L_, H_, B_), 256, 0, stream>>>(sig_ws, o_prior, o_s);

  proj_gemm_kernel<<<dim3(M / 16, 16), 128, 0, stream>>>(vn_ws, Wo, nullptr, o_out, M, D_, D_, 0);
}
